// NextGenSpatialGNNVulnerabilityDetector_61297773248667
// MI455X (gfx1250) — compile-verified, matrix operand order checked by reading, not run on hardware
//
#include <hip/hip_runtime.h>

// ---------------------------------------------------------------------------
// Fused GNN-centrality + 2-layer MLP for MI455X (gfx1250, wave32, WMMA).
//   deg  = scatter_add(ones over edge_index[1])
//   feats[r, j*64+q] = cent_j(r) * W_j[q] + b_j[q]           (built in LDS, bf16)
//   h    = relu(feats @ W1 + b1)                             (WMMA bf16 -> f32)
//   out  = h @ W2 + b2                                       (WMMA bf16 -> f32)
// W1/W2 are pre-packed once into the per-lane WMMA B-fragment order so each
// lane fetches its 16 bf16 K-values as one contiguous 32-byte load.
// ---------------------------------------------------------------------------

typedef __attribute__((ext_vector_type(8)))  float  v8f;
typedef __attribute__((ext_vector_type(16))) __bf16 v16bf;

#define HID 256
#define LDA 264   // padded LDS row stride (bf16 elems): 528B -> bank-friendly

__device__ __forceinline__ unsigned short f2bf(float f) {
  unsigned u = __builtin_bit_cast(unsigned, f);
  u += 0x7fffu + ((u >> 16) & 1u);          // round-to-nearest-even
  return (unsigned short)(u >> 16);
}

__global__ void k_zero(unsigned* __restrict__ deg, int n) {
  int i = blockIdx.x * blockDim.x + threadIdx.x;
  if (i < n) deg[i] = 0u;
}

__global__ void k_scatter(const int* __restrict__ col, unsigned* __restrict__ deg,
                          int n_edges) {
  int i = blockIdx.x * blockDim.x + threadIdx.x;
  if (i < n_edges) atomicAdd(&deg[col[i]], 1u);
}

// Repack row-major fp32 W[K=256][N=256] into bf16 B-fragments:
// P[(kt*2+half)*256 + n][16] holds, in VGPR0..7 pair order, the K-values that
// the lane (N=n, half) consumes for K-tile kt of v_wmma_f32_16x16x32_bf16.
__global__ void k_pack(const float* __restrict__ W1, const float* __restrict__ W2,
                       unsigned short* __restrict__ P1, unsigned short* __restrict__ P2) {
  int tid = blockIdx.x * blockDim.x + threadIdx.x;      // 8192 total
  const float* W = (tid < 4096) ? W1 : W2;
  unsigned short* P = (tid < 4096) ? P1 : P2;
  int r    = tid & 4095;
  int kt   = r >> 9;
  int half = (r >> 8) & 1;
  int n    = r & 255;
  unsigned short* dst = P + ((((kt << 1) + half) * 256 + n) << 4);
#pragma unroll
  for (int p = 0; p < 8; ++p) {
    int kb = kt * 32 + ((p < 4) ? (half * 8 + 2 * p) : (16 + half * 8 + 2 * (p - 4)));
    dst[2 * p]     = f2bf(W[kb * 256 + n]);
    dst[2 * p + 1] = f2bf(W[(kb + 1) * 256 + n]);
  }
}

__global__ __launch_bounds__(256)
void k_fused(const unsigned* __restrict__ deg, const float* __restrict__ noise,
             const float* __restrict__ Wd, const float* __restrict__ bd,
             const float* __restrict__ Wb, const float* __restrict__ bb,
             const float* __restrict__ Wc, const float* __restrict__ bc,
             const float* __restrict__ We, const float* __restrict__ be,
             const unsigned short* __restrict__ P1, const float* __restrict__ b1,
             const unsigned short* __restrict__ P2, const float* __restrict__ b2,
             float* __restrict__ out, int n_nodes) {
  __shared__ unsigned short sA[32 * LDA];   // feats tile, bf16
  __shared__ unsigned short sH[32 * LDA];   // relu(h) tile, bf16

  const int rowBase = blockIdx.x * 32;
  const float invN1 = 1.0f / (float)(n_nodes - 1);

  // ---- stage 1: build feats[32 x 256] (bf16) in LDS -----------------------
  {
    int t = threadIdx.x;
    int frow = t >> 3;            // 0..31
    int sub  = t & 7;             // 8 threads per row
    int gRow = rowBase + frow;
    if (gRow >= n_nodes) gRow = n_nodes - 1;   // tail clamp (reads only)
    float dc = (float)deg[gRow] * invN1;
    float cents[4];
    cents[0] = dc;
    cents[1] = dc * (1.0f + 0.1f * noise[gRow]);
    cents[2] = 1.0f / (dc + 1e-8f);
    cents[3] = sqrtf(dc);
    const float* Ws[4]  = {Wd, Wb, Wc, We};
    const float* Bsv[4] = {bd, bb, bc, be};
#pragma unroll
    for (int j = 0; j < 4; ++j) {
      float c = cents[j];
      const float* Wj = Ws[j];
      const float* Bj = Bsv[j];
#pragma unroll
      for (int p = 0; p < 4; ++p) {
        int k = (sub + p * 8) * 2;     // even col within the 64-wide segment
        float f0 = fmaf(c, Wj[k],     Bj[k]);
        float f1 = fmaf(c, Wj[k + 1], Bj[k + 1]);
        unsigned pk = (unsigned)f2bf(f0) | ((unsigned)f2bf(f1) << 16);
        *(unsigned*)&sA[frow * LDA + j * 64 + k] = pk;
      }
    }
  }
  __syncthreads();

  const int lane    = threadIdx.x & 31;
  const int wave    = threadIdx.x >> 5;
  const int waveRow = (wave >> 2) * 16;   // 0 | 16
  const int waveCol = (wave & 3) * 64;    // 0,64,128,192
  const int half    = lane >> 4;
  const int ml      = lane & 15;
  const v8f vzero   = {0.f, 0.f, 0.f, 0.f, 0.f, 0.f, 0.f, 0.f};

  // ---- stage 2: GEMM1  h = relu(feats @ W1 + b1) --------------------------
  v8f acc[4] = {vzero, vzero, vzero, vzero};
#pragma unroll
  for (int kt = 0; kt < 8; ++kt) {
    union { v16bf v; unsigned u[8]; } af;
    int rowL = waveRow + ml;
#pragma unroll
    for (int p = 0; p < 8; ++p) {
      int k = kt * 32 + ((p < 4) ? (half * 8 + 2 * p) : (16 + half * 8 + 2 * (p - 4)));
      af.u[p] = *(const unsigned*)&sA[rowL * LDA + k];
    }
#pragma unroll
    for (int nt = 0; nt < 4; ++nt) {
      int col = waveCol + nt * 16 + ml;
      v16bf bf;
      __builtin_memcpy(&bf, P1 + ((((kt << 1) + half) * 256 + col) << 4), 32);
      acc[nt] = __builtin_amdgcn_wmma_f32_16x16x32_bf16(
          false, af.v, false, bf, (short)0, acc[nt], false, false);
    }
  }
#pragma unroll
  for (int nt = 0; nt < 4; ++nt) {
    int col = waveCol + nt * 16 + ml;
    float bias = b1[col];
#pragma unroll
    for (int v = 0; v < 8; ++v) {
      float h = acc[nt][v] + bias;
      h = fmaxf(h, 0.0f);
      int r = waveRow + v + 8 * half;
      sH[r * LDA + col] = f2bf(h);
    }
  }
  __syncthreads();

  // ---- stage 3: GEMM2  out = h @ W2 + b2 ----------------------------------
  v8f acc2[4] = {vzero, vzero, vzero, vzero};
#pragma unroll
  for (int kt = 0; kt < 8; ++kt) {
    union { v16bf v; unsigned u[8]; } af;
    int rowL = waveRow + ml;
#pragma unroll
    for (int p = 0; p < 8; ++p) {
      int k = kt * 32 + ((p < 4) ? (half * 8 + 2 * p) : (16 + half * 8 + 2 * (p - 4)));
      af.u[p] = *(const unsigned*)&sH[rowL * LDA + k];
    }
#pragma unroll
    for (int nt = 0; nt < 4; ++nt) {
      int col = waveCol + nt * 16 + ml;
      v16bf bf;
      __builtin_memcpy(&bf, P2 + ((((kt << 1) + half) * 256 + col) << 4), 32);
      acc2[nt] = __builtin_amdgcn_wmma_f32_16x16x32_bf16(
          false, af.v, false, bf, (short)0, acc2[nt], false, false);
    }
  }
#pragma unroll
  for (int nt = 0; nt < 4; ++nt) {
    int col = waveCol + nt * 16 + ml;
    float bias = b2[col];
#pragma unroll
    for (int v = 0; v < 8; ++v) {
      int r = rowBase + waveRow + v + 8 * half;
      if (r < n_nodes) out[r * HID + col] = acc2[nt][v] + bias;
    }
  }
}

extern "C" void kernel_launch(void* const* d_in, const int* in_sizes, int n_in,
                              void* d_out, int out_size, void* d_ws, size_t ws_size,
                              hipStream_t stream) {
  // inputs: 0=x (unused), 1=edge_index, 2=noise, 3..10 = W/b deg,bet,clo,eig,
  //         11=W1, 12=b1, 13=W2, 14=b2
  const int*   edge_index = (const int*)d_in[1];
  const float* noise      = (const float*)d_in[2];
  const float* Wd = (const float*)d_in[3];
  const float* bd = (const float*)d_in[4];
  const float* Wb = (const float*)d_in[5];
  const float* bb = (const float*)d_in[6];
  const float* Wc = (const float*)d_in[7];
  const float* bc = (const float*)d_in[8];
  const float* We = (const float*)d_in[9];
  const float* be = (const float*)d_in[10];
  const float* W1 = (const float*)d_in[11];
  const float* b1 = (const float*)d_in[12];
  const float* W2 = (const float*)d_in[13];
  const float* b2 = (const float*)d_in[14];
  float* out = (float*)d_out;

  const int n_nodes = in_sizes[2];
  const int n_edges = in_sizes[1] / 2;

  // workspace: [deg u32 x n_nodes][W1 packed bf16 128KB][W2 packed bf16 128KB]
  char* ws = (char*)d_ws;
  unsigned* deg = (unsigned*)ws;
  size_t off = ((size_t)n_nodes * 4 + 1023) & ~(size_t)1023;
  unsigned short* P1 = (unsigned short*)(ws + off);
  unsigned short* P2 = (unsigned short*)(ws + off + (size_t)HID * HID * 2);

  k_zero<<<(n_nodes + 255) / 256, 256, 0, stream>>>(deg, n_nodes);
  k_scatter<<<(n_edges + 255) / 256, 256, 0, stream>>>(edge_index + n_edges, deg, n_edges);
  k_pack<<<8192 / 256, 256, 0, stream>>>(W1, W2, P1, P2);
  k_fused<<<(n_nodes + 31) / 32, 256, 0, stream>>>(
      deg, noise, Wd, bd, Wb, bb, Wc, bc, We, be,
      P1, b1, P2, b2, out, n_nodes);
}
